// ChunkedAttention_37374805409943
// MI455X (gfx1250) — compile-verified, matrix operand order checked by reading, not run on hardware
//
#include <hip/hip_runtime.h>
#include <hip/hip_bf16.h>
#include <math.h>

typedef __attribute__((ext_vector_type(16))) _Float16 v16h;
typedef __attribute__((ext_vector_type(8)))  _Float16 v8h;
typedef __attribute__((ext_vector_type(8)))  float    v8f;

#define S_LEN 4096
#define DMODEL 1024
#define NHEADS 16
#define HDIM 64
#define CHUNKLEN 1024

// ---------------------------------------------------------------------------
// WMMA helpers: v_wmma_f32_16x16x32_f16, D = A(16x32) * B(32x16) + C(16x16 f32)
// A layout (wave32): lane L (row M = L%16); halves 0..7 -> K = base+0..7,
// halves 8..15 -> K = base+16..23, base = (L>=16)?8:0.
// B layout mirrors A with N = L%16. C/D: VGPR r, lane L: M = r + ((L>=16)?8:0).
// ---------------------------------------------------------------------------
__device__ __forceinline__ v8f wmma_f16(v16h a, v16h b, v8f c) {
  return __builtin_amdgcn_wmma_f32_16x16x32_f16(
      false, a, false, b, (short)0, c, false, false);
}

__device__ __forceinline__ v16h pack16(v8h lo, v8h hi) {
  v16h r;
#pragma unroll
  for (int t = 0; t < 8; ++t) { r[t] = lo[t]; r[8 + t] = hi[t]; }
  return r;
}

__device__ __forceinline__ v16h load_frag_f16(const _Float16* __restrict__ row, int kb) {
  v8h lo = *(const v8h*)(row + kb);
  v8h hi = *(const v8h*)(row + kb + 16);
  return pack16(lo, hi);
}

// ---------------------------------------------------------------------------
// Kernel 1a: elementwise f32 -> f16
// ---------------------------------------------------------------------------
__global__ void cvt_x_kernel(const float* __restrict__ x, _Float16* __restrict__ xh,
                             int n) {
  int i = blockIdx.x * blockDim.x + threadIdx.x;
  if (i < n) xh[i] = (_Float16)x[i];
}

// ---------------------------------------------------------------------------
// Kernel 1b: convert + transpose weight  w[K][N] f32  ->  wT[N][K] f16
// ---------------------------------------------------------------------------
__global__ void cvtT_kernel(const float* __restrict__ w, _Float16* __restrict__ wT,
                            int K, int N) {
  int n = blockIdx.x * blockDim.x + threadIdx.x;
  int k = blockIdx.y;
  if (n < N) wT[(size_t)n * K + k] = (_Float16)w[(size_t)k * N + n];
}

// ---------------------------------------------------------------------------
// Kernel 2: QKV projection GEMM.  out = xh(f16 SxK) @ wT^T(f16) + bias
// Each wave computes a 32(M) x 64(N) tile (B fragments reused by 2 A frags).
// mode 0: out[h][s][d]; mode 1: out[h][d][s]
// ---------------------------------------------------------------------------
__global__ __launch_bounds__(256) void gemm_qkv_kernel(
    const _Float16* __restrict__ xh, const _Float16* __restrict__ wT,
    const float* __restrict__ bias, _Float16* __restrict__ out, int mode) {
  const int K = DMODEL;
  int wave   = (blockIdx.x * 256 + threadIdx.x) >> 5;   // 0..2047
  int lane16 = threadIdx.x & 15;
  int hi8    = ((threadIdx.x >> 4) & 1) << 3;           // 0 or 8
  int tile_m = wave & 127;                              // 128 tiles of 32 rows
  int tile_n = wave >> 7;                               // 16 tiles of 64 cols
  int m0 = tile_m * 32, n0 = tile_n * 64;

  v8f c0[4] = {}, c1[4] = {};
  const _Float16* arow0 = xh + (size_t)(m0 + lane16) * K;
  const _Float16* arow1 = arow0 + (size_t)16 * K;
  const _Float16* wrow  = wT + (size_t)(n0 + lane16) * K;

  for (int k0 = 0; k0 < K; k0 += 32) {
    int kb = k0 + hi8;
    v16h a0 = load_frag_f16(arow0, kb);
    v16h a1 = load_frag_f16(arow1, kb);
#pragma unroll
    for (int j = 0; j < 4; ++j) {
      v16h b = load_frag_f16(wrow + (size_t)(j * 16) * K, kb);
      c0[j] = wmma_f16(a0, b, c0[j]);
      c1[j] = wmma_f16(a1, b, c1[j]);
    }
  }

#pragma unroll
  for (int j = 0; j < 4; ++j) {
    int n = n0 + j * 16 + lane16;
    float bj = bias[n];
    int head = n >> 6, d = n & 63;
#pragma unroll
    for (int r = 0; r < 8; ++r) {
      int s0r = m0 + r + hi8;
      int s1r = s0r + 16;
      float v0 = c0[j][r] + bj;
      float v1 = c1[j][r] + bj;
      if (mode == 0) {
        out[((size_t)head * S_LEN + s0r) * HDIM + d] = (_Float16)v0;
        out[((size_t)head * S_LEN + s1r) * HDIM + d] = (_Float16)v1;
      } else {
        out[((size_t)head * HDIM + d) * S_LEN + s0r] = (_Float16)v0;
        out[((size_t)head * HDIM + d) * S_LEN + s1r] = (_Float16)v1;
      }
    }
  }
}

// ---------------------------------------------------------------------------
// Kernel 3: flash-style block-causal attention. One wave per (16-q-row tile, head).
// 64 keys per iteration: 8 score WMMAs + amortized softmax + 8 PV WMMAs.
// qh,kh: [H][S][64] f16;  vt: [H][64][S] f16;  aout: [S][H*64] f16
// ---------------------------------------------------------------------------
__global__ __launch_bounds__(32) void attn_kernel(
    const _Float16* __restrict__ qh, const _Float16* __restrict__ kh,
    const _Float16* __restrict__ vt, _Float16* __restrict__ aout) {
  __shared__ __align__(16) _Float16 sP[16][72];   // 16 q-rows x 64 keys (+pad)

  int qtile  = blockIdx.x;
  int head   = blockIdx.y;
  int lane16 = threadIdx.x & 15;
  int hi8    = ((threadIdx.x >> 4) & 1) << 3;
  int m0     = qtile * 16;
  int kv_end = ((m0 / CHUNKLEN) + 1) * CHUNKLEN;  // block-causal: full chunks
  const float scale = 0.125f;                     // 64^-0.5

  const _Float16* qrow = qh + ((size_t)head * S_LEN + m0 + lane16) * HDIM;
  v16h aq0 = load_frag_f16(qrow, hi8);        // d 0..31
  v16h aq1 = load_frag_f16(qrow + 32, hi8);   // d 32..63

  float rowmax[8], rowsum[8];
#pragma unroll
  for (int r = 0; r < 8; ++r) { rowmax[r] = -INFINITY; rowsum[r] = 0.0f; }
  v8f o[4] = {};

  for (int kb = 0; kb < kv_end; kb += 64) {
    // --- scores for 64 keys: four 16x16 tiles, Hd=64 split in two WMMAs each
    const _Float16* kr = kh + ((size_t)head * S_LEN + kb + lane16) * HDIM;
    union { v8f v; float f[8]; } u[4], av;
#pragma unroll
    for (int t = 0; t < 4; ++t) {
      const _Float16* krow = kr + (size_t)(t * 16) * HDIM;
      v8f s = {};
      s = wmma_f16(aq0, load_frag_f16(krow,      hi8), s);
      s = wmma_f16(aq1, load_frag_f16(krow + 32, hi8), s);
      u[t].v = s * scale;
    }

    // --- online softmax; each row lives across 16 lanes of a half-wave
#pragma unroll
    for (int r = 0; r < 8; ++r) {
      float mx = fmaxf(fmaxf(u[0].f[r], u[1].f[r]), fmaxf(u[2].f[r], u[3].f[r]));
      mx = fmaxf(mx, __shfl_xor(mx, 1));
      mx = fmaxf(mx, __shfl_xor(mx, 2));
      mx = fmaxf(mx, __shfl_xor(mx, 4));
      mx = fmaxf(mx, __shfl_xor(mx, 8));
      float nm    = fmaxf(rowmax[r], mx);
      float alpha = __expf(rowmax[r] - nm);
      rowmax[r]   = nm;
      float ps = 0.0f;
#pragma unroll
      for (int t = 0; t < 4; ++t) {
        float p = __expf(u[t].f[r] - nm);
        u[t].f[r] = p;
        ps += p;
      }
      ps += __shfl_xor(ps, 1);
      ps += __shfl_xor(ps, 2);
      ps += __shfl_xor(ps, 4);
      ps += __shfl_xor(ps, 8);
      rowsum[r] = rowsum[r] * alpha + ps;
      av.f[r]   = alpha;
    }
#pragma unroll
    for (int j = 0; j < 4; ++j) o[j] = o[j] * av.v;

    // --- redistribute P (C-layout) -> A-layout through LDS
    __syncthreads();
#pragma unroll
    for (int r = 0; r < 8; ++r) {
#pragma unroll
      for (int t = 0; t < 4; ++t)
        sP[r + hi8][t * 16 + lane16] = (_Float16)u[t].f[r];
    }
    __syncthreads();
    v16h ap0 = pack16(*(const v8h*)&sP[lane16][hi8],
                      *(const v8h*)&sP[lane16][hi8 + 16]);
    v16h ap1 = pack16(*(const v8h*)&sP[lane16][32 + hi8],
                      *(const v8h*)&sP[lane16][32 + hi8 + 16]);

    // --- O += P @ V   (V transposed: contiguous over keys per d)
    const _Float16* vbase = vt + ((size_t)head * HDIM + lane16) * S_LEN + kb;
#pragma unroll
    for (int j = 0; j < 4; ++j) {
      const _Float16* vr = vbase + (size_t)(j * 16) * S_LEN;
      o[j] = wmma_f16(ap0, load_frag_f16(vr,      hi8), o[j]);
      o[j] = wmma_f16(ap1, load_frag_f16(vr + 32, hi8), o[j]);
    }
  }

  union { v8f v; float f[8]; } iv;
#pragma unroll
  for (int r = 0; r < 8; ++r) iv.f[r] = 1.0f / rowsum[r];
#pragma unroll
  for (int j = 0; j < 4; ++j) {
    union { v8f v; float f[8]; } uo;
    uo.v = o[j] * iv.v;
#pragma unroll
    for (int r = 0; r < 8; ++r) {
      int s = m0 + r + hi8;
      aout[(size_t)s * DMODEL + head * HDIM + j * 16 + lane16] = (_Float16)uo.f[r];
    }
  }
}

// ---------------------------------------------------------------------------
// Kernel 4: output projection GEMM.  out(f32) = aout(f16) @ woT^T(f16) + bo
// 32(M) x 64(N) wave tiles.
// ---------------------------------------------------------------------------
__global__ __launch_bounds__(256) void gemm_out_kernel(
    const _Float16* __restrict__ ain, const _Float16* __restrict__ wT,
    const float* __restrict__ bias, float* __restrict__ out) {
  const int K = DMODEL, N = DMODEL;
  int wave   = (blockIdx.x * 256 + threadIdx.x) >> 5;
  int lane16 = threadIdx.x & 15;
  int hi8    = ((threadIdx.x >> 4) & 1) << 3;
  int tile_m = wave & 127;
  int tile_n = wave >> 7;
  int m0 = tile_m * 32, n0 = tile_n * 64;

  v8f c0[4] = {}, c1[4] = {};
  const _Float16* arow0 = ain + (size_t)(m0 + lane16) * K;
  const _Float16* arow1 = arow0 + (size_t)16 * K;
  const _Float16* wrow  = wT + (size_t)(n0 + lane16) * K;

  for (int k0 = 0; k0 < K; k0 += 32) {
    int kb = k0 + hi8;
    v16h a0 = load_frag_f16(arow0, kb);
    v16h a1 = load_frag_f16(arow1, kb);
#pragma unroll
    for (int j = 0; j < 4; ++j) {
      v16h b = load_frag_f16(wrow + (size_t)(j * 16) * K, kb);
      c0[j] = wmma_f16(a0, b, c0[j]);
      c1[j] = wmma_f16(a1, b, c1[j]);
    }
  }

#pragma unroll
  for (int j = 0; j < 4; ++j) {
    int n = n0 + j * 16 + lane16;
    float bj = bias[n];
#pragma unroll
    for (int r = 0; r < 8; ++r) {
      int s0r = m0 + r + hi8;
      out[(size_t)s0r * N + n]        = c0[j][r] + bj;
      out[(size_t)(s0r + 16) * N + n] = c1[j][r] + bj;
    }
  }
}

// ---------------------------------------------------------------------------
// Host launch
// ---------------------------------------------------------------------------
extern "C" void kernel_launch(void* const* d_in, const int* in_sizes, int n_in,
                              void* d_out, int out_size, void* d_ws, size_t ws_size,
                              hipStream_t stream) {
  const float* x  = (const float*)d_in[0];
  const float* wq = (const float*)d_in[1];
  const float* bq = (const float*)d_in[2];
  const float* wk = (const float*)d_in[3];
  const float* bk = (const float*)d_in[4];
  const float* wv = (const float*)d_in[5];
  const float* bv = (const float*)d_in[6];
  const float* wo = (const float*)d_in[7];
  const float* bo = (const float*)d_in[8];
  float* out = (float*)d_out;

  char* ws = (char*)d_ws;
  const size_t MB = 1024 * 1024;
  _Float16* wqT  = (_Float16*)(ws + 0 * MB);    // [1024][1024] f16, 2 MB each
  _Float16* wkT  = (_Float16*)(ws + 2 * MB);
  _Float16* wvT  = (_Float16*)(ws + 4 * MB);
  _Float16* woT  = (_Float16*)(ws + 6 * MB);
  _Float16* qh   = (_Float16*)(ws + 8 * MB);    // [16][4096][64], 8 MB each
  _Float16* kh   = (_Float16*)(ws + 16 * MB);
  _Float16* vt   = (_Float16*)(ws + 24 * MB);   // [16][64][4096]
  _Float16* aout = (_Float16*)(ws + 32 * MB);   // [4096][1024]
  _Float16* xh   = (_Float16*)(ws + 40 * MB);   // [4096][1024]

  int nx = S_LEN * DMODEL;
  cvt_x_kernel<<<(nx + 255) / 256, 256, 0, stream>>>(x, xh, nx);

  dim3 tgrid(DMODEL / 256, DMODEL);
  cvtT_kernel<<<tgrid, 256, 0, stream>>>(wq, wqT, DMODEL, DMODEL);
  cvtT_kernel<<<tgrid, 256, 0, stream>>>(wk, wkT, DMODEL, DMODEL);
  cvtT_kernel<<<tgrid, 256, 0, stream>>>(wv, wvT, DMODEL, DMODEL);
  cvtT_kernel<<<tgrid, 256, 0, stream>>>(wo, woT, DMODEL, DMODEL);

  // 2048 wave-tiles per GEMM, 8 waves per 256-thread block -> 256 blocks
  gemm_qkv_kernel<<<256, 256, 0, stream>>>(xh, wqT, bq, qh, 0);
  gemm_qkv_kernel<<<256, 256, 0, stream>>>(xh, wkT, bk, kh, 0);
  gemm_qkv_kernel<<<256, 256, 0, stream>>>(xh, wvT, bv, vt, 1);

  attn_kernel<<<dim3(S_LEN / 16, NHEADS), 32, 0, stream>>>(qh, kh, vt, aout);

  gemm_out_kernel<<<256, 256, 0, stream>>>(aout, woT, bo, out);
}
